// graphNetwork_faust_57389353009180
// MI455X (gfx1250) — compile-verified, matrix operand order checked by reading, not run on hardware
//
#include <hip/hip_runtime.h>
#include <math.h>

// ---------------------------------------------------------------------------
// Problem constants (from reference)
// ---------------------------------------------------------------------------
#define NN   8192      // nodes
#define EE   65536     // edges
#define NLAYER 8
#define HSTEP 0.1f

typedef __attribute__((ext_vector_type(16))) _Float16 v16h;
typedef __attribute__((ext_vector_type(8)))  float    v8f;
typedef __attribute__((ext_vector_type(8)))  _Float16 h8;   // 16 bytes
typedef __attribute__((ext_vector_type(2)))  _Float16 h2;   // 4 bytes

enum GemmMode { MODE_STORE = 0, MODE_STORE_STATS = 1, MODE_AXPY = 2,
                MODE_STORE_F16 = 3, MODE_BIAS_ELU_F16 = 4, MODE_BIAS_TSTORE = 5 };

// Async global->LDS copy of 16 bytes (CDNA5 GLOBAL_LOAD_ASYNC_TO_LDS_B128,
// tracked by ASYNCcnt). Inline asm: portable across ROCm7.2 / upstream clang.
__device__ __forceinline__ void async_copy_b128(const _Float16* g, _Float16* l)
{
    unsigned           lofs = (unsigned)(size_t)l;              // LDS byte offset
    unsigned long long ga   = (unsigned long long)(size_t)g;
    asm volatile("global_load_async_to_lds_b128 %0, %1, off"
                 :: "v"(lofs), "v"(ga) : "memory");
}
__device__ __forceinline__ void wait_async0()
{
    asm volatile("s_wait_asynccnt 0x0" ::: "memory");
}

// ---------------------------------------------------------------------------
// WMMA GEMM:  C[O, M] = A[O, K] * B[K, M]
// A, B are f16 in memory; accumulate f32 via v_wmma_f32_16x16x32_f16.
// Block: 256 threads = 8 waves; C-tile 64(O) x 128(M); K chunks of 32.
// A tile staged with async global->LDS b128 copies (layout-identical);
// B tile transposed into LDS [n][k] with paired-k b32 stores.
// Requires O%64==0, M%128==0, K%32==0 (true for all calls).
// ---------------------------------------------------------------------------
__global__ __launch_bounds__(256)
void gemm_wmma(const _Float16* __restrict__ A, const _Float16* __restrict__ B,
               float* __restrict__ C, _Float16* __restrict__ C16,
               int K, int M, int mode, const float* __restrict__ bias,
               float* __restrict__ stats, float hscale, int ldT)
{
    __shared__ __align__(16) _Float16 sA[64 * 32];    // [m][k], k contiguous
    __shared__ __align__(16) _Float16 sB[128 * 32];   // [n][k], k contiguous

    const int tid  = threadIdx.x;
    const int lane = tid & 31;
    const int wave = tid >> 5;
    const int wm   = wave & 3;        // 16-row group
    const int wn   = wave >> 2;       // 64-col group
    const int hsel = lane >> 4;       // lane half of wave32
    const int mrow = lane & 15;
    const int kofs = hsel * 8;        // per CDNA5 16-bit A/B fragment layout

    const int o0 = blockIdx.y * 64;
    const int m0 = blockIdx.x * 128;

    v8f acc[4];
#pragma unroll
    for (int t = 0; t < 4; ++t)
#pragma unroll
        for (int r = 0; r < 8; ++r) acc[t][r] = 0.0f;

    // staging index maps
    const int am  = tid >> 2;           // 0..63
    const int ak  = (tid & 3) * 8;      // 0,8,16,24
    const int bq  = tid >> 4;           // 0..15  -> k pair (2*bq, 2*bq+1)
    const int bn0 = (tid & 15) * 8;     // 0..120

    for (int k0 = 0; k0 < K; k0 += 32) {
        // ---- A tile (64x32 f16): async copy, LDS layout == memory layout
        const _Float16* ga = A + (size_t)(o0 + am) * K + (k0 + ak);
        async_copy_b128(ga, &sA[am * 32 + ak]);

        // ---- B tile (32x128 f16) -> LDS transposed [n][k]
        const _Float16* gb = B + (size_t)(k0 + 2 * bq) * M + (m0 + bn0);
        h8 r0 = *(const h8*)gb;
        h8 r1 = *(const h8*)(gb + M);
#pragma unroll
        for (int i = 0; i < 8; ++i) {
            h2 pr; pr[0] = r0[i]; pr[1] = r1[i];
            *(h2*)&sB[(bn0 + i) * 32 + 2 * bq] = pr;   // 4B store, aligned
        }

        if (k0 + 32 < K) {
            __builtin_prefetch(gb + 32, 0, 1);                 // next A chunk lives in L2 anyway
            __builtin_prefetch(gb + (size_t)32 * M, 0, 1);     // next B chunk
        }

        wait_async0();          // own wave's async LDS writes complete
        __syncthreads();        // all waves' staging visible

        // ---- A fragment: row = wm*16+mrow; halves kofs..+7 and kofs+16..+23
        const h8* pa = (const h8*)&sA[(wm * 16 + mrow) * 32 + kofs];
        h8 alo = pa[0];
        h8 ahi = pa[2];
        v16h a;
#pragma unroll
        for (int i = 0; i < 8; ++i) { a[i] = alo[i]; a[8 + i] = ahi[i]; }

#pragma unroll
        for (int t = 0; t < 4; ++t) {
            const h8* pb = (const h8*)&sB[(wn * 64 + t * 16 + mrow) * 32 + kofs];
            h8 blo = pb[0];
            h8 bhi = pb[2];
            v16h b;
#pragma unroll
            for (int i = 0; i < 8; ++i) { b[i] = blo[i]; b[8 + i] = bhi[i]; }
            acc[t] = __builtin_amdgcn_wmma_f32_16x16x32_f16(
                         false, a, false, b, (short)0, acc[t], false, false);
        }
        __syncthreads();
    }

    // ---- epilogue ----
    float s = 0.0f, s2 = 0.0f;
#pragma unroll
    for (int t = 0; t < 4; ++t) {
        const int col = m0 + wn * 64 + t * 16 + mrow;
#pragma unroll
        for (int r = 0; r < 8; ++r) {
            const int row = o0 + wm * 16 + hsel * 8 + r;    // C fragment layout
            float v = acc[t][r];
            const size_t idx = (size_t)row * M + col;
            if (mode == MODE_STORE) {
                C[idx] = v;
            } else if (mode == MODE_STORE_STATS) {
                C[idx] = v; s += v; s2 += v * v;
            } else if (mode == MODE_AXPY) {
                C[idx] = C[idx] + hscale * v;
            } else if (mode == MODE_STORE_F16) {
                C16[idx] = (_Float16)v;
            } else if (mode == MODE_BIAS_ELU_F16) {
                v += bias[row];
                v = v > 0.0f ? v : expm1f(v);
                C16[idx] = (_Float16)v;
            } else { // MODE_BIAS_TSTORE
                v += bias[row];
                C[(size_t)col * ldT + row] = v;
            }
        }
    }
    if (mode == MODE_STORE_STATS) {
#pragma unroll
        for (int off = 16; off > 0; off >>= 1) {
            s  += __shfl_xor(s,  off, 32);
            s2 += __shfl_xor(s2, off, 32);
        }
        if (lane == 0) { atomicAdd(&stats[0], s); atomicAdd(&stats[1], s2); }
    }
}

// ---------------------------------------------------------------------------
// Opening convs (K = 3 or 16) with fused LN statistics; f32 out (pre-LN).
// ---------------------------------------------------------------------------
__global__ __launch_bounds__(256)
void small_conv(const float* __restrict__ W, const float* __restrict__ X,
                float* __restrict__ Y, int K, int M, float* __restrict__ stats)
{
    const int id = blockIdx.x * 256 + threadIdx.x;
    const int o = id / M;
    const int m = id % M;
    float s = 0.0f;
    for (int k = 0; k < K; ++k) s += W[o * K + k] * X[(size_t)k * M + m];
    Y[(size_t)o * M + m] = s;
    float a = s, b = s * s;
#pragma unroll
    for (int off = 16; off > 0; off >>= 1) {
        a += __shfl_xor(a, off, 32);
        b += __shfl_xor(b, off, 32);
    }
    if ((threadIdx.x & 31) == 0) { atomicAdd(&stats[0], a); atomicAdd(&stats[1], b); }
}

// Global layer-norm (over ALL elements) + tanh; reads f32, writes f16.
__global__ __launch_bounds__(256)
void ln_tanh(const float* __restrict__ X, _Float16* __restrict__ Y,
             const float* __restrict__ stats, float invCount)
{
    const size_t id = (size_t)blockIdx.x * 256 + threadIdx.x;
    const float mean = stats[0] * invCount;
    const float var  = stats[1] * invCount - mean * mean;
    const float rstd = rsqrtf(var + 1e-5f);
    Y[id] = (_Float16)tanhf((X[id] - mean) * rstd);
}

// f32 -> f16 conversion
__global__ __launch_bounds__(256)
void cvt_f16(const float* __restrict__ src, _Float16* __restrict__ dst, int n)
{
    const int i = blockIdx.x * 256 + threadIdx.x;
    if (i < n) dst[i] = (_Float16)src[i];
}

// dxe_cat[384, E] = [nodeAve(xn); xe; nodeGrad(xn)]  (f16 output for GEMM B)
__global__ __launch_bounds__(256)
void gather_cat_e(const float* __restrict__ xn, const float* __restrict__ xe,
                  const int* __restrict__ I, const int* __restrict__ J,
                  _Float16* __restrict__ cat)
{
    const int id = blockIdx.x * 256 + threadIdx.x;  // 128 * E
    const int c = id >> 16;
    const int e = id & (EE - 1);
    const int i = I[e], j = J[e];
    const float xi = xn[(size_t)c * NN + i];
    const float xj = xn[(size_t)c * NN + j];
    cat[(size_t)c * EE + e]         = (_Float16)(0.5f * (xi + xj));
    cat[(size_t)(128 + c) * EE + e] = (_Float16)xe[(size_t)c * EE + e];
    cat[(size_t)(256 + c) * EE + e] = (_Float16)(xi - xj);
}

// dxn_cat[384, N] (f32): rows 0..255 zeroed for scatter, rows 256..383 = xn
__global__ __launch_bounds__(256)
void init_cat_n(const float* __restrict__ xn, float* __restrict__ cat)
{
    const int id = blockIdx.x * 256 + threadIdx.x;  // 384 * N
    const int c = id >> 13;
    const int n = id & (NN - 1);
    cat[id] = (c < 256) ? 0.0f : xn[(size_t)(c - 256) * NN + n];
}

// rows 0..127 += edgeAve(xe); rows 128..255 += edgeDiv(xe)  (f32 atomics)
__global__ __launch_bounds__(256)
void scatter_e(const float* __restrict__ xe, const int* __restrict__ I,
               const int* __restrict__ J, float* __restrict__ cat)
{
    const int id = blockIdx.x * 256 + threadIdx.x;  // 128 * E
    const int c = id >> 16;
    const int e = id & (EE - 1);
    const float g = xe[(size_t)c * EE + e];
    const int i = I[e], j = J[e];
    atomicAdd(&cat[(size_t)c * NN + i],          0.5f * g);
    atomicAdd(&cat[(size_t)c * NN + j],          0.5f * g);
    atomicAdd(&cat[(size_t)(128 + c) * NN + i],  g);
    atomicAdd(&cat[(size_t)(128 + c) * NN + j], -g);
}

// log_softmax in-place over each row of [NN, NN]
__global__ __launch_bounds__(256)
void log_softmax_rows(float* __restrict__ X)
{
    __shared__ float red[256];
    float* row = X + (size_t)blockIdx.x * NN;
    const int tid = threadIdx.x;

    float mx = -INFINITY;
    for (int i = tid; i < NN; i += 256) mx = fmaxf(mx, row[i]);
    red[tid] = mx; __syncthreads();
    for (int s = 128; s > 0; s >>= 1) {
        if (tid < s) red[tid] = fmaxf(red[tid], red[tid + s]);
        __syncthreads();
    }
    mx = red[0]; __syncthreads();

    float sum = 0.0f;
    for (int i = tid; i < NN; i += 256) sum += expf(row[i] - mx);
    red[tid] = sum; __syncthreads();
    for (int s = 128; s > 0; s >>= 1) {
        if (tid < s) red[tid] += red[tid + s];
        __syncthreads();
    }
    const float lse = mx + logf(red[0]);
    for (int i = tid; i < NN; i += 256) row[i] = row[i] - lse;
}

__global__ void zero_stats(float* __restrict__ s) { s[threadIdx.x] = 0.0f; }

__global__ void sig_alpha(const float* __restrict__ a, float* __restrict__ o)
{
    o[0] = 1.0f / (1.0f + expf(-a[0]));
}

// ---------------------------------------------------------------------------
// Host orchestration
// ---------------------------------------------------------------------------
extern "C" void kernel_launch(void* const* d_in, const int* in_sizes, int n_in,
                              void* d_out, int out_size, void* d_ws, size_t ws_size,
                              hipStream_t stream)
{
    const float* xn_in    = (const float*)d_in[0];
    const float* xe_in    = (const float*)d_in[1];
    const float* K1Nopen  = (const float*)d_in[2];
    const float* K2Nopen  = (const float*)d_in[3];
    const float* K1Eopen  = (const float*)d_in[4];
    const float* K2Eopen  = (const float*)d_in[5];
    const float* KNclose  = (const float*)d_in[6];
    const float* KE1      = (const float*)d_in[7];
    const float* KE2      = (const float*)d_in[8];
    const float* KN1      = (const float*)d_in[9];
    const float* KN2      = (const float*)d_in[10];
    const float* lin1_w   = (const float*)d_in[11];
    const float* lin1_b   = (const float*)d_in[12];
    const float* lin2_w   = (const float*)d_in[13];
    const float* lin2_b   = (const float*)d_in[14];
    const float* alpha    = (const float*)d_in[15];
    const int*   Iidx     = (const int*)d_in[16];
    const int*   Jidx     = (const int*)d_in[17];

    float* out = (float*)d_out;
    float* ws  = (float*)d_ws;

    // ---- f32 region ----
    size_t off = 0;
    float* xn    = ws + off; off += (size_t)128 * NN;   // state
    float* xe    = ws + off; off += (size_t)128 * EE;   // state
    float* hid_e = ws + off; off += (size_t)256 * EE;   // pre-LN edge hidden
    float* hid_n = ws + off; off += (size_t)256 * NN;   // pre-LN node hidden (+opening tmp)
    float* cat_n = ws + off; off += (size_t)384 * NN;   // f32 scatter target
    float* stats = ws + off; off += 64;

    // ---- f16 arena ----
    _Float16* f16a = (_Float16*)(ws + off);
    size_t h = 0;
    _Float16* cat_e16  = f16a + h; h += (size_t)384 * EE;
    _Float16* hte16    = f16a + h; h += (size_t)256 * EE;   // post-LN edge (+opening)
    _Float16* cat_n16  = f16a + h; h += (size_t)384 * NN;   // also xn16 at close
    _Float16* htn16    = f16a + h; h += (size_t)256 * NN;   // post-LN node (+opening)
    _Float16* close16  = f16a + h; h += (size_t)128 * NN;
    _Float16* lin1o16  = f16a + h; h += (size_t)256 * NN;
    _Float16* wK2N     = f16a + h; h += 128 * 128;
    _Float16* wK2E     = f16a + h; h += 128 * 128;
    _Float16* wKNc     = f16a + h; h += 128 * 128;
    _Float16* wKE1     = f16a + h; h += (size_t)NLAYER * 256 * 384;
    _Float16* wKE2     = f16a + h; h += (size_t)NLAYER * 128 * 256;
    _Float16* wKN1     = f16a + h; h += (size_t)NLAYER * 256 * 384;
    _Float16* wKN2     = f16a + h; h += (size_t)NLAYER * 128 * 256;
    _Float16* wL1      = f16a + h; h += 256 * 128;
    _Float16* wL2      = f16a + h; h += (size_t)NN * 256;

    auto cvt = [&](const float* s, _Float16* d, int n) {
        cvt_f16<<<(n + 255) / 256, 256, 0, stream>>>(s, d, n);
    };
    auto gemm = [&](const _Float16* A, const _Float16* B, float* C, _Float16* C16,
                    int O, int K, int M, int mode,
                    const float* bias, float* st, float hs, int ldT) {
        dim3 grid(M / 128, O / 64);
        gemm_wmma<<<grid, 256, 0, stream>>>(A, B, C, C16, K, M, mode, bias, st, hs, ldT);
    };

    zero_stats<<<1, 64, 0, stream>>>(stats);

    // ---- one-time weight conversion to f16 ----
    cvt(K2Nopen, wK2N, 128 * 128);
    cvt(K2Eopen, wK2E, 128 * 128);
    cvt(KNclose, wKNc, 128 * 128);
    cvt(KE1, wKE1, NLAYER * 256 * 384);
    cvt(KE2, wKE2, NLAYER * 128 * 256);
    cvt(KN1, wKN1, NLAYER * 256 * 384);
    cvt(KN2, wKN2, NLAYER * 128 * 256);
    cvt(lin1_w, wL1, 256 * 128);
    cvt(lin2_w, wL2, NN * 256);

    // ---- opening: node ----
    small_conv<<<(128 * NN) / 256, 256, 0, stream>>>(K1Nopen, xn_in, hid_n, 3, NN, stats + 0);
    ln_tanh<<<(128 * NN) / 256, 256, 0, stream>>>(hid_n, htn16, stats + 0, 1.0f / (128.0f * NN));
    gemm(wK2N, htn16, xn, nullptr, 128, 128, NN, MODE_STORE, nullptr, nullptr, 0.f, 0);

    // ---- opening: edge ----
    small_conv<<<(128 * EE) / 256, 256, 0, stream>>>(K1Eopen, xe_in, hid_e, 16, EE, stats + 2);
    ln_tanh<<<(128 * EE) / 256, 256, 0, stream>>>(hid_e, hte16, stats + 2, 1.0f / (128.0f * EE));
    gemm(wK2E, hte16, xe, nullptr, 128, 128, EE, MODE_STORE, nullptr, nullptr, 0.f, 0);

    // ---- layer loop ----
    for (int l = 0; l < NLAYER; ++l) {
        float* stE = stats + 2 * (2 + 2 * l);
        float* stN = stats + 2 * (3 + 2 * l);
        const _Float16* ke1 = wKE1 + (size_t)l * 256 * 384;
        const _Float16* ke2 = wKE2 + (size_t)l * 128 * 256;
        const _Float16* kn1 = wKN1 + (size_t)l * 256 * 384;
        const _Float16* kn2 = wKN2 + (size_t)l * 128 * 256;

        // edge update: xe += H * K2 tanh(LN(K1 [ave; xe; grad]))
        gather_cat_e<<<(128 * EE) / 256, 256, 0, stream>>>(xn, xe, Iidx, Jidx, cat_e16);
        gemm(ke1, cat_e16, hid_e, nullptr, 256, 384, EE, MODE_STORE_STATS, nullptr, stE, 0.f, 0);
        ln_tanh<<<(256 * EE) / 256, 256, 0, stream>>>(hid_e, hte16, stE, 1.0f / (256.0f * EE));
        gemm(ke2, hte16, xe, nullptr, 128, 256, EE, MODE_AXPY, nullptr, nullptr, HSTEP, 0);

        // node update: xn += H * K2 tanh(LN(K1 [edgeAve; edgeDiv; xn]))
        init_cat_n<<<(384 * NN) / 256, 256, 0, stream>>>(xn, cat_n);
        scatter_e<<<(128 * EE) / 256, 256, 0, stream>>>(xe, Iidx, Jidx, cat_n);
        cvt(cat_n, cat_n16, 384 * NN);
        gemm(kn1, cat_n16, hid_n, nullptr, 256, 384, NN, MODE_STORE_STATS, nullptr, stN, 0.f, 0);
        ln_tanh<<<(256 * NN) / 256, 256, 0, stream>>>(hid_n, htn16, stN, 1.0f / (256.0f * NN));
        gemm(kn2, htn16, xn, nullptr, 128, 256, NN, MODE_AXPY, nullptr, nullptr, HSTEP, 0);
    }

    // ---- close + classifier ----
    _Float16* xn16 = cat_n16;                       // reuse (free after layers)
    cvt(xn, xn16, 128 * NN);
    gemm(wKNc, xn16, nullptr, close16, 128, 128, NN, MODE_STORE_F16, nullptr, nullptr, 0.f, 0);
    gemm(wL1, close16, nullptr, lin1o16, 256, 128, NN, MODE_BIAS_ELU_F16, lin1_b, nullptr, 0.f, 0);
    gemm(wL2, lin1o16, out, nullptr, NN, 256, NN, MODE_BIAS_TSTORE, lin2_b, nullptr, 0.f, NN);

    log_softmax_rows<<<NN, 256, 0, stream>>>(out);
    sig_alpha<<<1, 1, 0, stream>>>(alpha, out + (size_t)NN * NN);
}
